// forget_1_88012469829692
// MI455X (gfx1250) — matrix operand order repaired and checked by reading, not compiled, not run on hardware
//
#include <hip/hip_runtime.h>

// ---------------------------------------------------------------------------
// Problem constants (T,B,R,P,H) = (256, 32, 64, 64, 256)
// rows = B*R = 2048, gate width H = 256, fused K = P + H = 320
// ---------------------------------------------------------------------------
#define TT 256
#define BB 32
#define RR 64
#define PP 64
#define HH 256
#define ROWS (BB * RR)           // 2048
#define XEL  (TT * BB * RR * PP) // 33554432 elements per (T,B,R,P) tensor
#define HEL  (ROWS * HH)         // 524288

typedef _Float16 f16;
typedef __attribute__((ext_vector_type(16))) _Float16 v16h;
typedef __attribute__((ext_vector_type(8)))  _Float16 v8h;
typedef __attribute__((ext_vector_type(8)))  float    v8f;

#define WMMA(a, b, c) \
  __builtin_amdgcn_wmma_f32_16x16x32_f16(false, (a), false, (b), (short)0, (c), false, false)

static __device__ __forceinline__ v16h make16(v8h lo, v8h hi) {
  return __builtin_shufflevector(lo, hi, 0,1,2,3,4,5,6,7,8,9,10,11,12,13,14,15);
}

// A fragment, 16x32 f16, row-major source with given stride.
// Lane l: row m = row0 + (l&15). hi = l>>4.
//   elements 0..15 -> K = kbase + hi*16 + e       (32 contiguous bytes)
static __device__ __forceinline__ v16h load_a_frag(const f16* base, int stride,
                                                   int row0, int kbase, int lane) {
  int m  = row0 + (lane & 15);
  int hi = lane >> 4;
  const f16* p = base + (size_t)m * stride + kbase + hi * 16;
  v8h lo = *(const v8h*)(p);
  v8h hh = *(const v8h*)(p + 8);
  return make16(lo, hh);
}

// B fragment from a pre-packed buffer: 32 contiguous bytes per lane (coalesced
// b128 pairs). Packed layout: buf[frag*512 + lane*16 + e] where element e of
// lane l corresponds to B[K = frag_kbase + (l>>4)*16 + e][N = frag_n0 + (l&15)].
static __device__ __forceinline__ v16h load_b_packed(const f16* wp, int frag, int lane) {
  const f16* p = wp + ((size_t)frag << 9) + (lane << 4);
  v8h lo = *(const v8h*)(p);
  v8h hh = *(const v8h*)(p + 8);
  return make16(lo, hh);
}

// ---------------------------------------------------------------------------
// Pack gate weights [W_i* (64xH) ; W_h* (HxH)] (K=320) and Wn (64x64) into
// per-lane B-fragment layout:  wpack[((g*16+ct)*10+kc)*512 + lane*16 + e]
// ---------------------------------------------------------------------------
__global__ void pack_weights(const float* Wn,
                             const float* W_ii, const float* W_hi,
                             const float* W_if, const float* W_hf,
                             const float* W_ig, const float* W_hg,
                             const float* W_io, const float* W_ho,
                             f16* wpack, f16* wnpack) {
  int idx = blockIdx.x * blockDim.x + threadIdx.x;
  const int NGATE = 4 * 16 * 10 * 512;  // 327680
  if (idx < NGATE) {
    int e    = idx & 15;
    int lane = (idx >> 4) & 31;
    int rest = idx >> 9;          // (g*16+ct)*10 + kc
    int kc   = rest % 10;
    int gct  = rest / 10;
    int ct   = gct & 15;
    int g    = gct >> 4;
    int K    = kc * 32 + (lane >> 4) * 16 + e;
    int col  = ct * 16 + (lane & 15);
    const float* Wi = (g == 0) ? W_ii : (g == 1) ? W_if : (g == 2) ? W_ig : W_io;
    const float* Wh = (g == 0) ? W_hi : (g == 1) ? W_hf : (g == 2) ? W_hg : W_ho;
    float v = (K < PP) ? Wi[K * HH + col] : Wh[(K - PP) * HH + col];
    wpack[idx] = (f16)v;
  } else if (idx < NGATE + 4 * 2 * 512) {
    int i2   = idx - NGATE;
    int e    = i2 & 15;
    int lane = (i2 >> 4) & 31;
    int rest = i2 >> 9;           // ct*2 + kc
    int kc   = rest & 1;
    int ct   = rest >> 1;
    int K    = kc * 32 + (lane >> 4) * 16 + e;
    int col  = ct * 16 + (lane & 15);
    wnpack[i2] = (f16)Wn[K * PP + col];
  }
}

// ---------------------------------------------------------------------------
// Transpose x (f16 row-major, per (t,b) a 64x64 tile) into B-fragment layout:
// xpack[(tb*8 + ct*2 + kc)*512 + lane*16 + e] = x[tb][kc*32+(lane>>4)*16+e][ct*16+(lane&15)]
// One memory-bound pass; writes fully coalesced, reads L2-tiled (8KB tiles).
// ---------------------------------------------------------------------------
__global__ void pack_x_bfrag(const f16* __restrict__ x16, f16* __restrict__ xpack,
                             size_t n) {
  size_t i      = (size_t)blockIdx.x * blockDim.x + threadIdx.x;
  size_t stride = (size_t)gridDim.x * blockDim.x;
  for (; i < n; i += stride) {
    int e       = (int)(i & 15);
    int lane    = (int)((i >> 4) & 31);
    size_t rest = i >> 9;                // tb*8 + ct*2 + kc
    int kc      = (int)(rest & 1);
    int ct      = (int)((rest >> 1) & 3);
    size_t tb   = rest >> 3;
    int K       = kc * 32 + (lane >> 4) * 16 + e;
    int nn      = ct * 16 + (lane & 15);
    xpack[i]    = x16[tb * 4096 + (size_t)K * 64 + nn];
  }
}

__global__ void cvt_f32_f16(const float* __restrict__ s, f16* __restrict__ d, size_t n) {
  size_t i      = (size_t)blockIdx.x * blockDim.x + threadIdx.x;
  size_t stride = (size_t)gridDim.x * blockDim.x;
  for (; i < n; i += stride) d[i] = (f16)s[i];
}

__global__ void copy_f32(const float* __restrict__ s, float* __restrict__ d, size_t n) {
  size_t i      = (size_t)blockIdx.x * blockDim.x + threadIdx.x;
  size_t stride = (size_t)gridDim.x * blockDim.x;
  for (; i < n; i += stride) d[i] = s[i];
}

// ---------------------------------------------------------------------------
// ax_t = (a_t @ x_t) @ Wn + bn   for all (t,b): 8192 blocks, 8 waves each.
// Two chained 64x64x64 WMMA GEMMs, staging tmp in LDS as f16.
// Both B operands now come from packed fragment buffers (b128 loads only).
// ---------------------------------------------------------------------------
__global__ void __launch_bounds__(256)
ax_kernel(const f16* __restrict__ a16, const f16* __restrict__ xpack,
          const f16* __restrict__ wnpack, const float* __restrict__ bn,
          f16* __restrict__ ax16) {
  const int tb = blockIdx.x;
  const f16* A  = a16 + (size_t)tb * 4096;
  const f16* XP = xpack + (size_t)tb * 4096;
  f16* OUT      = ax16 + (size_t)tb * 4096;

  __shared__ f16 tmp[64 * 64];

  const int tid  = threadIdx.x;
  const int wave = tid >> 5;
  const int lane = tid & 31;
  const int hi   = lane >> 4;
  const int n    = lane & 15;

  // GEMM1: tmp = a_t @ x_t
#pragma unroll
  for (int s = 0; s < 2; ++s) {
    int tile = wave * 2 + s;
    int rt = tile >> 2, ct = tile & 3;
    v8f acc = {0.f, 0.f, 0.f, 0.f, 0.f, 0.f, 0.f, 0.f};
#pragma unroll
    for (int kc = 0; kc < 2; ++kc) {
      v16h af = load_a_frag(A, 64, rt * 16, kc * 32, lane);
      v16h bf = load_b_packed(XP, ct * 2 + kc, lane);
      acc = WMMA(af, bf, acc);
    }
#pragma unroll
    for (int r = 0; r < 8; ++r)
      tmp[(rt * 16 + hi * 8 + r) * 64 + ct * 16 + n] = (f16)acc[r];
  }
  __syncthreads();

  // GEMM2: out = tmp @ Wn + bn
#pragma unroll
  for (int s = 0; s < 2; ++s) {
    int tile = wave * 2 + s;
    int rt = tile >> 2, ct = tile & 3;
    v8f acc = {0.f, 0.f, 0.f, 0.f, 0.f, 0.f, 0.f, 0.f};
#pragma unroll
    for (int kc = 0; kc < 2; ++kc) {
      v16h af = load_a_frag(tmp, 64, rt * 16, kc * 32, lane);
      v16h bf = load_b_packed(wnpack, ct * 2 + kc, lane);
      acc = WMMA(af, bf, acc);
    }
    int col = ct * 16 + n;
    float b = bn[col];
#pragma unroll
    for (int r = 0; r < 8; ++r)
      OUT[(rt * 16 + hi * 8 + r) * 64 + col] = (f16)(acc[r] + b);
  }
}

// ---------------------------------------------------------------------------
// One recurrent step. 128 blocks x 256 threads; block owns 16 rows.
// Waves: w>>1 = gate (0=i,1=f,2=g,3=o), w&1 = column half (8 tiles of 16).
// Fused GEMM: [x_t | h] (16x320) @ [[W_i*];[W_h*]] (320x256) per gate,
// then sigmoid()@w reductions (waves 0-3,6-7) and c/h update (waves 4-5).
// ---------------------------------------------------------------------------
__global__ void __launch_bounds__(256)
step_kernel(const f16* __restrict__ xg, const f16* __restrict__ axg,
            f16* __restrict__ h16, float* __restrict__ c,
            const f16* __restrict__ wpack,
            const float* __restrict__ b_ii, const float* __restrict__ b_hi,
            const float* __restrict__ b_if, const float* __restrict__ b_hf,
            const float* __restrict__ b_ig, const float* __restrict__ b_hg,
            const float* __restrict__ b_io, const float* __restrict__ b_ho,
            const float* __restrict__ w, float* __restrict__ out_t) {
  const int row0 = blockIdx.x * 16;
  const int tid  = threadIdx.x;
  const int wave = tid >> 5;
  const int lane = tid & 31;
  const int hi   = lane >> 4;
  const int n    = lane & 15;
  const int gate = wave >> 1;
  const int half = wave & 1;

  __shared__ float red[6][16];

  const f16* asrc = (gate == 1) ? axg : xg;  // K<64 comes from x_t (or ax_t for f)

  v8f acc[8];
#pragma unroll
  for (int j = 0; j < 8; ++j)
    acc[j] = (v8f){0.f, 0.f, 0.f, 0.f, 0.f, 0.f, 0.f, 0.f};

  // 10 K-chunks of 32: kc 0..1 -> input part (stride 64), kc 2..9 -> h (stride 256)
#pragma unroll
  for (int kc = 0; kc < 10; ++kc) {
    int kbase = kc * 32;
    v16h af = (kbase < PP)
                  ? load_a_frag(asrc, PP, row0, kbase, lane)
                  : load_a_frag(h16, HH, row0, kbase - PP, lane);
#pragma unroll
    for (int j = 0; j < 8; ++j) {
      int ct  = half * 8 + j;
      v16h bf = load_b_packed(wpack, (gate * 16 + ct) * 10 + kc, lane);
      acc[j]  = WMMA(af, bf, acc[j]);
    }
  }

  const float* bi = (gate == 0) ? b_ii : (gate == 1) ? b_if : (gate == 2) ? b_ig : b_io;
  const float* bh = (gate == 0) ? b_hi : (gate == 1) ? b_hf : (gate == 2) ? b_hg : b_ho;

  if (gate != 2) {
    // i/f/o: sigmoid then dot with w, reduced over all 256 columns per row.
    float part[8];
#pragma unroll
    for (int r = 0; r < 8; ++r) part[r] = 0.f;
#pragma unroll
    for (int j = 0; j < 8; ++j) {
      int col    = (half * 8 + j) * 16 + n;
      float bias = bi[col] + bh[col];
      float wc   = w[col];
#pragma unroll
      for (int r = 0; r < 8; ++r) {
        float v = acc[j][r] + bias;
        float s = 1.0f / (1.0f + __expf(-v));
        part[r] += s * wc;
      }
    }
    // reduce across the 16 lanes of this half (wave32 xor shuffles)
#pragma unroll
    for (int off = 1; off < 16; off <<= 1)
#pragma unroll
      for (int r = 0; r < 8; ++r)
        part[r] += __shfl_xor(part[r], off, 32);
    int slot = (gate == 0) ? half : (gate == 1) ? 2 + half : 4 + half;
    if (n == 0)
#pragma unroll
      for (int r = 0; r < 8; ++r)
        red[slot][hi * 8 + r] = part[r];
  } else {
    // g gate: tanh, keep full matrix in the accumulators.
#pragma unroll
    for (int j = 0; j < 8; ++j) {
      int col    = (half * 8 + j) * 16 + n;
      float bias = bi[col] + bh[col];
#pragma unroll
      for (int r = 0; r < 8; ++r)
        acc[j][r] = tanhf(acc[j][r] + bias);
    }
  }
  __syncthreads();

  if (gate == 2) {
    // waves 4,5 update c and h for their 16 rows x 128 cols.
#pragma unroll
    for (int j = 0; j < 8; ++j) {
      int col = (half * 8 + j) * 16 + n;
#pragma unroll
      for (int r = 0; r < 8; ++r) {
        int rl      = hi * 8 + r;
        int row     = row0 + rl;
        float i_s   = red[0][rl] + red[1][rl];
        float f_s   = red[2][rl] + red[3][rl];
        float o_s   = red[4][rl] + red[5][rl];
        size_t idx  = (size_t)row * HH + col;
        float c_new = f_s * c[idx] + i_s * acc[j][r];
        c[idx]      = c_new;
        float h_new = o_s * tanhf(c_new);
        out_t[idx]  = h_new;
        h16[idx]    = (f16)h_new;
      }
    }
  }
}

// ---------------------------------------------------------------------------
// Host side
// ---------------------------------------------------------------------------
extern "C" void kernel_launch(void* const* d_in, const int* in_sizes, int n_in,
                              void* d_out, int out_size, void* d_ws, size_t ws_size,
                              hipStream_t stream) {
  (void)in_sizes; (void)n_in; (void)out_size; (void)ws_size;

  const float* x0  = (const float*)d_in[0];
  const float* x1  = (const float*)d_in[1];
  const float* h0  = (const float*)d_in[2];
  const float* c0  = (const float*)d_in[3];
  const float* W   = (const float*)d_in[4];   // (H,1) -> w vector
  const float* Wn  = (const float*)d_in[5];
  const float* bn  = (const float*)d_in[6];
  const float* W_ii = (const float*)d_in[7];
  const float* W_hi = (const float*)d_in[8];
  const float* b_ii = (const float*)d_in[9];
  const float* b_hi = (const float*)d_in[10];
  const float* W_if = (const float*)d_in[11];
  const float* W_hf = (const float*)d_in[12];
  const float* b_if = (const float*)d_in[13];
  const float* b_hf = (const float*)d_in[14];
  const float* W_ig = (const float*)d_in[15];
  const float* W_hg = (const float*)d_in[16];
  const float* b_ig = (const float*)d_in[17];
  const float* b_hg = (const float*)d_in[18];
  const float* W_io = (const float*)d_in[19];
  const float* W_ho = (const float*)d_in[20];
  const float* b_io = (const float*)d_in[21];
  const float* b_ho = (const float*)d_in[22];

  // Workspace carve (~272 MB)
  f16* x16    = (f16*)d_ws;              // XEL
  f16* a16    = x16 + XEL;               // XEL
  f16* ax16   = a16 + XEL;               // XEL
  f16* xpack  = ax16 + XEL;              // XEL (B-fragment layout of x)
  f16* h16    = xpack + XEL;             // HEL
  float* c    = (float*)(h16 + HEL);     // HEL
  f16* wpack  = (f16*)(c + HEL);         // 327680
  f16* wnpack = wpack + 327680;          // 4096

  // Pack weights into WMMA B-fragment layout.
  pack_weights<<<(327680 + 4096 + 255) / 256, 256, 0, stream>>>(
      Wn, W_ii, W_hi, W_if, W_hf, W_ig, W_hg, W_io, W_ho, wpack, wnpack);

  // Convert inputs to f16 / copy state.
  cvt_f32_f16<<<2048, 256, 0, stream>>>(x0, x16, (size_t)XEL);
  cvt_f32_f16<<<2048, 256, 0, stream>>>(x1, a16, (size_t)XEL);
  cvt_f32_f16<<<512, 256, 0, stream>>>(h0, h16, (size_t)HEL);
  copy_f32<<<512, 256, 0, stream>>>(c0, c, (size_t)HEL);

  // Transpose x into B-fragment layout for the a_t @ x_t GEMM.
  pack_x_bfrag<<<4096, 256, 0, stream>>>(x16, xpack, (size_t)XEL);

  // Precompute ax_t for all (t,b) with WMMA.
  ax_kernel<<<TT * BB, 256, 0, stream>>>(a16, xpack, wnpack, bn, ax16);

  // Sequential recurrence: 256 step launches.
  float* out = (float*)d_out;
  for (int t = 0; t < TT; ++t) {
    step_kernel<<<ROWS / 16, 256, 0, stream>>>(
        x16 + (size_t)t * (BB * RR * PP), ax16 + (size_t)t * (BB * RR * PP),
        h16, c, wpack,
        b_ii, b_hi, b_if, b_hf, b_ig, b_hg, b_io, b_ho,
        W, out + (size_t)t * HEL);
  }
}
